// MambaLayer_9646496547077
// MI455X (gfx1250) — compile-verified, matrix operand order checked by reading, not run on hardware
//
#include <hip/hip_runtime.h>

// ---------------------------------------------------------------------------
// Mamba layer forward for MI455X (gfx1250): bf16 WMMA GEMMs + fused scan.
// B=2, L=2048, d_model=2048, d_inner=4096, d_state=16, d_conv=4, dt_rank=128
// ---------------------------------------------------------------------------

typedef __bf16 bf16_t;
typedef __attribute__((ext_vector_type(16))) bf16_t v16bf;
typedef __attribute__((ext_vector_type(8)))  bf16_t v8bf;
typedef __attribute__((ext_vector_type(8)))  float  v8f;

#define NB   2
#define LQ   2048
#define DMQ  2048
#define DIQ  4096
#define DSQ  16
#define DCQ  4
#define DTRQ 128
#define E2Q  (2 * DIQ)        // 8192
#define XDQ  (DTRQ + 2*DSQ)   // 160
#define XDP  256              // padded N for x_proj weights (valid-load trick)

// --------------------------- f32 -> bf16 cast ------------------------------
__global__ void cast_f32_bf16(const float* __restrict__ in,
                              bf16_t* __restrict__ out, long n) {
  long i = (long)blockIdx.x * blockDim.x + threadIdx.x;
  long stride = (long)gridDim.x * blockDim.x;
  for (; i < n; i += stride) out[i] = (bf16_t)in[i];
}

// --------------------------- WMMA fragment load ----------------------------
// 16-bit A-matrix lane layout (ISA 7.12.2): lane m (0..15) holds K = {0..7,
// 16..23}; lane m+16 holds K = {8..15, 24..31}. Two b128 loads per fragment.
__device__ __forceinline__ v16bf load_frag(const bf16_t* __restrict__ p) {
  v8bf lo = *(const v8bf*)p;
  v8bf hi = *(const v8bf*)(p + 16);
  v16bf f;
#pragma unroll
  for (int i = 0; i < 8; ++i) { f[i] = lo[i]; f[i + 8] = hi[i]; }
  return f;
}

// --------------------------- bf16 GEMM (f32 acc) ---------------------------
// C[M x N] (f32, row-major, ldc) = A[M x K] (bf16, row-major, lda)
//                                * B[N x K]^T (bf16, row-major, ldb)
// Block: 256 threads = 8 waves as 2(M) x 4(N); wave tile 64x64 via 4x4
// accumulators of v_wmma_f32_16x16x32_bf16. Requirements:
//   M % 128 == 0, K % 64 == 0, K >= 64.
//   B operand must have >= ceil(N,256) valid-to-READ rows (pad the buffer);
//   out-of-range rows feed only accumulator columns that are never stored.
// Double-buffered fragments: loads for tile k+32 issue before WMMAs of tile k.
__global__ __launch_bounds__(256, 1)
void gemm_bf16f32(const bf16_t* __restrict__ A, int lda, long long strideA,
                  const bf16_t* __restrict__ Bw, int ldb,
                  float* __restrict__ C, int ldc, long long strideC,
                  int N, int K) {
  const int lane = threadIdx.x & 31;
  const int wave = threadIdx.x >> 5;
  const int row0 = blockIdx.y * 128 + (wave >> 2) * 64;
  const int col0 = blockIdx.x * 256 + (wave & 3) * 64;
  const bf16_t* __restrict__ Ab = A + (long long)blockIdx.z * strideA;
  float* __restrict__ Cb = C + (long long)blockIdx.z * strideC;

  const int mlan = lane & 15;
  const int kh = (lane >> 4) << 3;   // 0 or 8

  const bf16_t* pa[4];
  const bf16_t* pb[4];
#pragma unroll
  for (int i = 0; i < 4; ++i) {
    pa[i] = Ab + (size_t)(row0 + i * 16 + mlan) * lda + kh;
    pb[i] = Bw + (size_t)(col0 + i * 16 + mlan) * ldb + kh;
  }

  v8f acc[4][4] = {};
  v16bf a0[4], b0[4], a1[4], b1[4];

  auto ldset = [&](v16bf (&af)[4], v16bf (&bf)[4], int k) {
#pragma unroll
    for (int i = 0; i < 4; ++i) af[i] = load_frag(pa[i] + k);
#pragma unroll
    for (int i = 0; i < 4; ++i) bf[i] = load_frag(pb[i] + k);
  };
  auto mm = [&](v16bf (&af)[4], v16bf (&bf)[4]) {
#pragma unroll
    for (int i = 0; i < 4; ++i)
#pragma unroll
      for (int j = 0; j < 4; ++j)
        acc[i][j] = __builtin_amdgcn_wmma_f32_16x16x32_bf16(
            false, af[i], false, bf[j], (short)0, acc[i][j], false, false);
  };

  ldset(a0, b0, 0);
  int k = 0;
  for (; k + 64 < K; k += 64) {
    ldset(a1, b1, k + 32);   // prefetch next half-tile
    mm(a0, b0);
    ldset(a0, b0, k + 64);   // prefetch tile after that
    mm(a1, b1);
  }
  ldset(a1, b1, k + 32);
  mm(a0, b0);
  mm(a1, b1);

  // C/D layout: VGPR r, lanes 0-15 -> M = r, lanes 16-31 -> M = r + 8;
  // N = lane & 15.  Store guard only (cheap, outside the K loop).
  const int nl = lane & 15;
  const int mo = (lane >> 4) << 3;
#pragma unroll
  for (int i = 0; i < 4; ++i) {
#pragma unroll
    for (int j = 0; j < 4; ++j) {
      const int cc = col0 + j * 16 + nl;
      if (cc < N) {
#pragma unroll
        for (int r = 0; r < 8; ++r)
          Cb[(size_t)(row0 + i * 16 + mo + r) * ldc + cc] = acc[i][j][r];
      }
    }
  }
}

// ------------------- depthwise causal conv1d + SiLU -------------------------
// x = xz[:, :, 0:DI] in (B, L, 8192); outputs fp32 (scan input) + bf16 (GEMM).
__global__ void conv_silu(const float* __restrict__ xz,
                          const float* __restrict__ cw,   // (DI, 4)
                          const float* __restrict__ cb,   // (DI)
                          float* __restrict__ xc, bf16_t* __restrict__ xcb) {
  long idx = (long)blockIdx.x * blockDim.x + threadIdx.x;
  const long total = (long)NB * LQ * DIQ;
  if (idx >= total) return;
  const int d = (int)(idx % DIQ);
  const long bl = idx / DIQ;
  const int l = (int)(bl % LQ);
  const int b = (int)(bl / LQ);
  float acc = cb[d];
#pragma unroll
  for (int k = 0; k < DCQ; ++k) {
    const int li = l - (DCQ - 1) + k;
    if (li >= 0)
      acc += cw[d * DCQ + k] * xz[((size_t)b * LQ + li) * E2Q + d];
  }
  const float s = acc / (1.0f + __expf(-acc));   // silu
  xc[idx] = s;
  xcb[idx] = (bf16_t)s;
}

// --------------------------- selective scan ---------------------------------
// One thread per (b, d) channel; 16 fp32 states in registers; sequential in t.
// Fuses: softplus(dt + bias), h-recurrence, y = h.C + D*u, SiLU(z) gating,
// bf16 output for the out-proj WMMA GEMM.
__global__ __launch_bounds__(256, 1)
void selective_scan(const float* __restrict__ xc,     // u     (B, L, DI)
                    const float* __restrict__ dtraw,  //       (B, L, DI)
                    const float* __restrict__ xdbl,   //       (B, L, 160)
                    const float* __restrict__ xz,     // z at [.., DI + d]
                    const float* __restrict__ A_log,  // (DI, 16)
                    const float* __restrict__ Dp,     // (DI)
                    const float* __restrict__ dtb,    // (DI)
                    bf16_t* __restrict__ ybf) {       //       (B, L, DI)
  const int d = blockIdx.x * blockDim.x + threadIdx.x;
  const int b = blockIdx.y;

  float a[DSQ], h[DSQ];
#pragma unroll
  for (int n = 0; n < DSQ; ++n) {
    a[n] = -__expf(A_log[(size_t)d * DSQ + n]);
    h[n] = 0.0f;
  }
  const float Dd = Dp[d];
  const float bias = dtb[d];

  for (int t = 0; t < LQ; ++t) {
    const size_t row = (size_t)b * LQ + t;
    const float u = xc[row * DIQ + d];
    const float v = dtraw[row * DIQ + d] + bias;
    const float delta = (v > 20.0f) ? v : log1pf(__expf(v));  // softplus

    float bt[2 * DSQ];
    const float4* p4 = (const float4*)(xdbl + row * XDQ + DTRQ);
#pragma unroll
    for (int q = 0; q < 8; ++q) ((float4*)bt)[q] = p4[q];

    const float du = delta * u;
    float y = 0.0f;
#pragma unroll
    for (int n = 0; n < DSQ; ++n) {
      const float dA = __expf(delta * a[n]);
      h[n] = h[n] * dA + du * bt[n];
      y += h[n] * bt[DSQ + n];
    }
    y += u * Dd;
    const float zz = xz[row * E2Q + DIQ + d];
    y *= zz / (1.0f + __expf(-zz));               // * silu(z)
    ybf[row * DIQ + d] = (bf16_t)y;
  }
}

// --------------------------------- launch -----------------------------------
extern "C" void kernel_launch(void* const* d_in, const int* in_sizes, int n_in,
                              void* d_out, int out_size, void* d_ws,
                              size_t ws_size, hipStream_t stream) {
  (void)in_sizes; (void)n_in; (void)out_size; (void)ws_size;
  const float* hs   = (const float*)d_in[0];  // (B, L, DM)
  const float* Win  = (const float*)d_in[1];  // (8192, 2048)
  const float* cw   = (const float*)d_in[2];  // (DI, 4)
  const float* cb   = (const float*)d_in[3];  // (DI)
  const float* Wxp  = (const float*)d_in[4];  // (160, DI)
  const float* Wdt  = (const float*)d_in[5];  // (DI, 128)
  const float* dtb  = (const float*)d_in[6];  // (DI)
  const float* Alog = (const float*)d_in[7];  // (DI, 16)
  const float* Dp   = (const float*)d_in[8];  // (DI)
  const float* Wout = (const float*)d_in[9];  // (DM, DI)
  float* out = (float*)d_out;                 // (B, L, DM)

  char* ws = (char*)d_ws;
  auto alloc = [&](size_t bytes) -> char* {
    char* p = ws;
    ws += (bytes + 255) & ~(size_t)255;
    return p;
  };
  bf16_t* hsb   = (bf16_t*)alloc((size_t)NB * LQ * DMQ * 2);
  bf16_t* Winb  = (bf16_t*)alloc((size_t)E2Q * DMQ * 2);
  bf16_t* Wxpb  = (bf16_t*)alloc((size_t)XDP * DIQ * 2);  // padded to 256 rows
  bf16_t* Wdtb  = (bf16_t*)alloc((size_t)DIQ * DTRQ * 2);
  bf16_t* Woutb = (bf16_t*)alloc((size_t)DMQ * DIQ * 2);
  float*  xzf   = (float*) alloc((size_t)NB * LQ * E2Q * 4);
  float*  xcf   = (float*) alloc((size_t)NB * LQ * DIQ * 4);
  bf16_t* xcb   = (bf16_t*)alloc((size_t)NB * LQ * DIQ * 2);
  float*  xdbl  = (float*) alloc((size_t)NB * LQ * XDQ * 4);
  bf16_t* xdblb = (bf16_t*)alloc((size_t)NB * LQ * XDQ * 2);
  float*  dtraw = (float*) alloc((size_t)NB * LQ * DIQ * 4);
  bf16_t* ybf   = (bf16_t*)alloc((size_t)NB * LQ * DIQ * 2);

  auto cast = [&](const float* src, bf16_t* dst, long n) {
    long g = (n + 255) / 256;
    if (g > 65535) g = 65535;
    cast_f32_bf16<<<dim3((unsigned)g), dim3(256), 0, stream>>>(src, dst, n);
  };
  cast(hs,   hsb,   (long)NB * LQ * DMQ);
  cast(Win,  Winb,  (long)E2Q * DMQ);
  cast(Wxp,  Wxpb,  (long)XDQ * DIQ);
  cast(Wdt,  Wdtb,  (long)DIQ * DTRQ);
  cast(Wout, Woutb, (long)DMQ * DIQ);

  // 1) in_proj: xz[b] (L x 8192) = hs[b] (L x 2048) * Win^T
  {
    dim3 g(E2Q / 256, LQ / 128, NB);
    gemm_bf16f32<<<g, 256, 0, stream>>>(hsb, DMQ, (long long)LQ * DMQ,
                                        Winb, DMQ,
                                        xzf, E2Q, (long long)LQ * E2Q,
                                        E2Q, DMQ);
  }
  // 2) depthwise causal conv + SiLU on x = xz[:, :, 0:DI]
  {
    const long total = (long)NB * LQ * DIQ;
    conv_silu<<<dim3((unsigned)((total + 255) / 256)), dim3(256), 0, stream>>>(
        xzf, cw, cb, xcf, xcb);
  }
  // 3) x_proj: xdbl[b] (L x 160) = xc[b] (L x 4096) * Wxp^T (padded reads)
  {
    dim3 g(XDP / 256, LQ / 128, NB);
    gemm_bf16f32<<<g, 256, 0, stream>>>(xcb, DIQ, (long long)LQ * DIQ,
                                        Wxpb, DIQ,
                                        xdbl, XDQ, (long long)LQ * XDQ,
                                        XDQ, DIQ);
  }
  cast(xdbl, xdblb, (long)NB * LQ * XDQ);
  // 4) dt_proj: dtraw[b] (L x 4096) = xdbl[b, :, 0:128] (L x 128) * Wdt^T
  {
    dim3 g(DIQ / 256, LQ / 128, NB);
    gemm_bf16f32<<<g, 256, 0, stream>>>(xdblb, XDQ, (long long)LQ * XDQ,
                                        Wdtb, DTRQ,
                                        dtraw, DIQ, (long long)LQ * DIQ,
                                        DIQ, DTRQ);
  }
  // 5) selective scan (fused softplus, D-skip, SiLU(z) gating) -> y bf16
  {
    dim3 g(DIQ / 256, NB);
    selective_scan<<<g, 256, 0, stream>>>(xcf, dtraw, xdbl, xzf, Alog, Dp, dtb,
                                          ybf);
  }
  // 6) out_proj: out[b] (L x 2048) = y[b] (L x 4096) * Wout^T
  {
    dim3 g(DMQ / 256, LQ / 128, NB);
    gemm_bf16f32<<<g, 256, 0, stream>>>(ybf, DIQ, (long long)LQ * DIQ,
                                        Woutb, DIQ,
                                        out, DMQ, (long long)LQ * DMQ,
                                        DMQ, DIQ);
  }
}
